// GCN_45509473469204
// MI455X (gfx1250) — matrix operand order repaired and checked by reading, not compiled, not run on hardware
//
#include <hip/hip_runtime.h>
#include <math.h>

// CDNA5 WMMA vector types (wave32)
typedef __attribute__((ext_vector_type(16))) _Float16 v16h;
typedef __attribute__((ext_vector_type(8)))  float    v8f;

// ---------------------------------------------------------------------------
// init: deg = 1 (self loop), agg1[N*16] = 0, agg2[N*2] = 0
// ---------------------------------------------------------------------------
__global__ void gcn_init_kernel(unsigned* __restrict__ deg,
                                float* __restrict__ agg1,
                                float* __restrict__ agg2, int N) {
    long long i = (long long)blockIdx.x * blockDim.x + threadIdx.x;
    if (i < (long long)N * 16) agg1[i] = 0.0f;
    if (i < (long long)N * 2)  agg2[i] = 0.0f;
    if (i < N)                 deg[i]  = 1u;
}

// ---------------------------------------------------------------------------
// degree count: deg[row[e]] += 1
// ---------------------------------------------------------------------------
__global__ void gcn_deg_kernel(const int* __restrict__ row,
                               unsigned* __restrict__ deg, int E) {
    int e = blockIdx.x * blockDim.x + threadIdx.x;
    if (e < E) atomicAdd(&deg[row[e]], 1u);
}

// ---------------------------------------------------------------------------
// dinv[i] = rsqrt(deg[i])  (deg >= 1 always: self loops)
// ---------------------------------------------------------------------------
__global__ void gcn_dinv_kernel(const unsigned* __restrict__ deg,
                                float* __restrict__ dinv, int N) {
    int i = blockIdx.x * blockDim.x + threadIdx.x;
    if (i < N) dinv[i] = rsqrtf((float)deg[i]);
}

// ---------------------------------------------------------------------------
// Y[m, 0:YCOLS] = dinv[m] * (X[m, 0:KD] @ W[KD, WCOLS]), via WMMA f16->f32.
// One wave = one 16-row tile, K padded to 32 with zeros (compile-time).
// A layout (16-bit A 16x32, ISA 7.12.2): lane l: M=l&15, g=l>>4;
//   g=0: elems 0..7 -> K=0..7,  elems 8..15 -> K=16..23
//   g=1: elems 0..7 -> K=8..15, elems 8..15 -> K=24..31
// B layout (16-bit B 32x16): lane l: N=l&15, g=l>>4; same K mapping as A
// C/D layout: elem r -> (M = r + 8*g, N = l&15)
// ---------------------------------------------------------------------------
template <int LDX, int KD, int WCOLS, int YCOLS>
__global__ void gcn_xw_wmma_kernel(const float* __restrict__ X,
                                   const float* __restrict__ W,
                                   const float* __restrict__ dinv,
                                   float* __restrict__ Y,
                                   int ntiles, int nrows) {
    int wave = blockIdx.x * (blockDim.x >> 5) + (threadIdx.x >> 5);
    if (wave >= ntiles) return;                 // wave-uniform: EXEC all-1 below
    int lane = threadIdx.x & 31;
    int g    = lane >> 4;
    int mn   = lane & 15;                       // M (for A) / N (for B, C)
    int m0   = wave * 16;

    // ---- pack A: row (m0+mn) of X, f32 -> f16; compile-time zero padding ----
    int arow = m0 + mn;
    const float* xrow = X + (size_t)(arow < nrows ? arow : nrows - 1) * LDX;
    v16h a;
    if (g == 0) {
#pragma unroll
        for (int e = 0; e < 8; ++e)             // K = e
            a[e] = (_Float16)((e < KD) ? xrow[e] : 0.0f);
#pragma unroll
        for (int e = 8; e < 16; ++e)            // K = e + 8
            a[e] = (_Float16)(((e + 8) < KD) ? xrow[e + 8] : 0.0f);
    } else {
#pragma unroll
        for (int e = 0; e < 8; ++e)             // K = e + 8
            a[e] = (_Float16)(((e + 8) < KD) ? xrow[e + 8] : 0.0f);
#pragma unroll
        for (int e = 8; e < 16; ++e)            // K = e + 16
            a[e] = (_Float16)(((e + 16) < KD) ? xrow[e + 16] : 0.0f);
    }

    // ---- pack B: column mn of W; clamp column, select-zero pad ----
    int wc = (WCOLS < 16 && mn >= WCOLS) ? 0 : mn;   // in-bounds address
    v16h b;
#pragma unroll
    for (int e = 0; e < 16; ++e) {
        int klo = (e < 8) ? e : e + 8;          // g == 0
        int khi = (e < 8) ? e + 8 : e + 16;     // g == 1
        float lo = (klo < KD) ? W[klo * WCOLS + wc] : 0.0f;
        float hi = (khi < KD) ? W[khi * WCOLS + wc] : 0.0f;
        float wv = (g == 0) ? lo : hi;
        if (WCOLS < 16 && mn >= WCOLS) wv = 0.0f;
        b[e] = (_Float16)wv;
    }

    v8f c = {};
    // (neg_a, A, neg_b, B, c_mod, C, reuse_a, reuse_b)
    c = __builtin_amdgcn_wmma_f32_16x16x32_f16(false, a, false, b,
                                               (short)0, c, false, false);

    // ---- epilogue: scale row m by dinv[m], store live columns ----
    if (m0 + 16 <= nrows) {                     // wave-uniform fast path
#pragma unroll
        for (int r = 0; r < 8; ++r) {
            int m = m0 + r + 8 * g;
            float s = dinv[m];
            if (YCOLS == 16 || mn < YCOLS)
                Y[(size_t)m * YCOLS + mn] = c[r] * s;
        }
    } else {
#pragma unroll
        for (int r = 0; r < 8; ++r) {
            int m = m0 + r + 8 * g;
            if (m < nrows && mn < YCOLS)
                Y[(size_t)m * YCOLS + mn] = c[r] * dinv[m];
        }
    }
}

// ---------------------------------------------------------------------------
// scatter-add: one thread per (edge, feature). AGG[row[e]*C + j] += Y[col[e]*C + j]
// C = 1<<SHIFT (16 for layer 1, 2 for layer 2). Coalesced gathers + 1 atomic/thread.
// ---------------------------------------------------------------------------
template <int SHIFT>
__global__ void gcn_scatter_kernel(const int* __restrict__ row,
                                   const int* __restrict__ col,
                                   const float* __restrict__ Y,
                                   float* __restrict__ AGG,
                                   long long total) {
    long long idx = (long long)blockIdx.x * blockDim.x + threadIdx.x;
    if (idx >= total) return;
    int       j = (int)(idx & ((1 << SHIFT) - 1));
    long long e = idx >> SHIFT;
    // keep the index stream ahead of the consumers (global_prefetch_b8)
    if ((threadIdx.x & 255) == 0) {
        __builtin_prefetch(col + e + 8192, 0, 1);
        __builtin_prefetch(row + e + 8192, 0, 1);
    }
    int r = row[e];
    int c = col[e];
    atomicAdd(&AGG[((size_t)r << SHIFT) + j], Y[((size_t)c << SHIFT) + j]);
}

// ---------------------------------------------------------------------------
// finalize layer 1 (in place on AGG1 -> H): h = relu((agg + y_self) * dinv + b)
// float4-vectorized: one thread per 4 features, N*4 threads.
// ---------------------------------------------------------------------------
__global__ void gcn_fin1_kernel(const float4* __restrict__ Y1,
                                float4* __restrict__ AGG1,
                                const float* __restrict__ dinv,
                                const float4* __restrict__ b1, int N) {
    long long idx = (long long)blockIdx.x * blockDim.x + threadIdx.x;
    if (idx >= (long long)N * 4) return;
    int i = (int)(idx >> 2);
    int q = (int)(idx & 3);
    float  s  = dinv[i];
    float4 ag = AGG1[idx];
    float4 ys = Y1[idx];
    float4 bb = b1[q];
    float4 o;
    o.x = fmaxf((ag.x + ys.x) * s + bb.x, 0.0f);
    o.y = fmaxf((ag.y + ys.y) * s + bb.y, 0.0f);
    o.z = fmaxf((ag.z + ys.z) * s + bb.z, 0.0f);
    o.w = fmaxf((ag.w + ys.w) * s + bb.w, 0.0f);
    AGG1[idx] = o;
}

// ---------------------------------------------------------------------------
// finalize layer 2 + log_softmax over the 2 classes
// ---------------------------------------------------------------------------
__global__ void gcn_fin2_kernel(const float* __restrict__ Y2,
                                const float* __restrict__ AGG2,
                                const float* __restrict__ dinv,
                                const float* __restrict__ b2,
                                float* __restrict__ out, int N) {
    int i = blockIdx.x * blockDim.x + threadIdx.x;
    if (i >= N) return;
    float s  = dinv[i];
    float z0 = (AGG2[2 * i + 0] + Y2[2 * i + 0]) * s + b2[0];
    float z1 = (AGG2[2 * i + 1] + Y2[2 * i + 1]) * s + b2[1];
    float m   = fmaxf(z0, z1);
    float lse = m + logf(expf(z0 - m) + expf(z1 - m));
    out[2 * i + 0] = z0 - lse;
    out[2 * i + 1] = z1 - lse;
}

static inline size_t gcn_align256(size_t v) { return (v + 255) & ~(size_t)255; }

extern "C" void kernel_launch(void* const* d_in, const int* in_sizes, int n_in,
                              void* d_out, int out_size, void* d_ws, size_t ws_size,
                              hipStream_t stream) {
    const float* x  = (const float*)d_in[0];   // [N, 20]
    const int*   ei = (const int*)d_in[1];     // [2, E] (JAX default x64-off -> int32)
    const float* W1 = (const float*)d_in[2];   // [20, 16]
    const float* b1 = (const float*)d_in[3];   // [16]
    const float* W2 = (const float*)d_in[4];   // [16, 2]
    const float* b2 = (const float*)d_in[5];   // [2]

    const int N = in_sizes[0] / 20;
    const int E = in_sizes[1] / 2;
    const int* row = ei;        // targets
    const int* col = ei + E;    // sources

    // workspace layout
    char*  ws  = (char*)d_ws;
    size_t off = 0;
    unsigned* deg  = (unsigned*)(ws + off); off = gcn_align256(off + (size_t)N * 4);
    float*    dinv = (float*)(ws + off);    off = gcn_align256(off + (size_t)N * 4);
    float*    y1   = (float*)(ws + off);    off = gcn_align256(off + (size_t)N * 16 * 4);
    float*    agg1 = (float*)(ws + off);    off = gcn_align256(off + (size_t)N * 16 * 4);
    float*    y2   = (float*)(ws + off);    off = gcn_align256(off + (size_t)N * 2 * 4);
    float*    agg2 = (float*)(ws + off);    off = gcn_align256(off + (size_t)N * 2 * 4);
    float*    out  = (float*)d_out;
    (void)ws_size; (void)n_in; (void)out_size;

    const int B = 256;                               // 8 waves / block (wave32)
    long long nf = (long long)N * 16;
    int nfBlocks = (int)((nf + B - 1) / B);

    // 1) init deg/agg buffers (every call: buffers are poisoned by harness)
    gcn_init_kernel<<<nfBlocks, B, 0, stream>>>(deg, agg1, agg2, N);
    // 2) degree count, then dinv = rsqrt(deg)
    gcn_deg_kernel<<<(E + B - 1) / B, B, 0, stream>>>(row, deg, E);
    gcn_dinv_kernel<<<(N + B - 1) / B, B, 0, stream>>>(deg, dinv, N);

    int ntiles = (N + 15) / 16;
    int wpb    = B / 32;
    int gB     = (ntiles + wpb - 1) / wpb;

    // 3) y1 = dinv .* (X @ W1)   [WMMA, K=20 padded to 32]
    gcn_xw_wmma_kernel<20, 20, 16, 16><<<gB, B, 0, stream>>>(x, W1, dinv, y1, ntiles, N);
    // 4) agg1[row] += y1[col]    (6.4M edges x 16 feats)
    long long s1 = (long long)E * 16;
    gcn_scatter_kernel<4><<<(int)((s1 + B - 1) / B), B, 0, stream>>>(row, col, y1, agg1, s1);
    // 5) h = relu((agg1 + y1) * dinv + b1)   (in place: agg1 -> h), float4 wide
    long long n4 = (long long)N * 4;
    gcn_fin1_kernel<<<(int)((n4 + B - 1) / B), B, 0, stream>>>(
        (const float4*)y1, (float4*)agg1, dinv, (const float4*)b1, N);

    // 6) y2 = dinv .* (h @ W2)   [WMMA, K=16 padded to 32, 2 live columns]
    gcn_xw_wmma_kernel<16, 16, 2, 2><<<gB, B, 0, stream>>>(agg1, W2, dinv, y2, ntiles, N);
    // 7) agg2[row] += y2[col]    (6.4M edges x 2 feats)
    long long s2 = (long long)E * 2;
    gcn_scatter_kernel<1><<<(int)((s2 + B - 1) / B), B, 0, stream>>>(row, col, y2, agg2, s2);
    // 8) out = log_softmax((agg2 + y2) * dinv + b2)
    gcn_fin2_kernel<<<(N + B - 1) / B, B, 0, stream>>>(y2, agg2, dinv, b2, out, N);
}
